// EncoderTransformer_9766755631602
// MI455X (gfx1250) — matrix operand order repaired and checked by reading, not compiled, not run on hardware
//
#include <hip/hip_runtime.h>
#include <hip/hip_bf16.h>
#include <math.h>

// ---------------------------------------------------------------------------
// Types for CDNA5 WMMA (wave32): v_wmma_f32_16x16x32_bf16
// ---------------------------------------------------------------------------
typedef __bf16 bf16_t;
typedef bf16_t bf16x8  __attribute__((ext_vector_type(8)));
typedef bf16_t bf16x16 __attribute__((ext_vector_type(16)));
typedef float  floatx8 __attribute__((ext_vector_type(8)));

// Let the backend pick a HW convert if gfx1250 has one (RNE either way).
static __device__ inline bf16_t f2bf(float f) {
    return static_cast<bf16_t>(f);
}
// Two floats -> packed dword of bf16 (candidates for v_cvt_pk_bf16_f32).
static __device__ inline unsigned int f2bf2(float lo, float hi) {
    unsigned short ua = __builtin_bit_cast(unsigned short, static_cast<bf16_t>(lo));
    unsigned short ub = __builtin_bit_cast(unsigned short, static_cast<bf16_t>(hi));
    return (unsigned int)ua | ((unsigned int)ub << 16);
}

#define LDSS 48   // LDS row stride in bf16 elems (96B: 16B-aligned fragments)

// ---------------------------------------------------------------------------
// Generic batched GEMM: C[M,N] = A[M,K] * op(B) + bias, fp32 in/out,
// bf16 WMMA compute with f32 accumulation, software-pipelined:
//   global->regs prefetch (next K-step) overlapped with WMMA (current),
//   double-buffered LDS, one barrier per K-step.
//   A: row-major, leading dim lda (elements). transB=0: B is KxN (ldb);
//   transB=1: B is NxK (ldb). Batch via blockIdx.z with element strides.
// Requires M%128==0, N%128==0, K%32==0, lda/ldb/ldc%4==0.
// ---------------------------------------------------------------------------
__global__ __launch_bounds__(256) void gemm_wmma_bf16(
    const float* __restrict__ A, const float* __restrict__ B,
    const float* __restrict__ bias, float* __restrict__ C,
    int M, int N, int K, int lda, int ldb, int ldc,
    long long strideA, long long strideB, long long strideC,
    int transB, int relu)
{
    __shared__ bf16_t lds_a[2][128 * LDSS];
    __shared__ bf16_t lds_b[2][128 * LDSS];

    const int tid = threadIdx.x;
    A += (long long)blockIdx.z * strideA;
    B += (long long)blockIdx.z * strideB;
    C += (long long)blockIdx.z * strideC;

    const int bm = blockIdx.y * 128;
    const int bn = blockIdx.x * 128;

    const int lane = tid & 31;
    const int wave = tid >> 5;
    const int wm   = (wave >> 2) * 64;   // 0 / 64
    const int wn   = (wave & 3) * 32;    // 0,32,64,96
    const int g    = lane >> 4;          // lane group (0/1)
    const int l16  = lane & 15;

    // staging registers (prefetched global data for one K-step)
    float4 ra[4], rb[4];

    const int tr  = tid >> 3;            // 0..31
    const int tc  = (tid & 7) * 4;       // 0..28
    const int tr2 = tid >> 5;            // 0..7   (k) for NN B staging
    const int tc2 = (tid & 31) * 4;      // 0..124 (n)

    // issue all 8 global_load_b128 back-to-back; no waits here
    auto load_tiles = [&](int kb) {
        #pragma unroll
        for (int r0 = 0; r0 < 4; ++r0)
            ra[r0] = *(const float4*)(A + (long long)(bm + r0 * 32 + tr) * lda + kb + tc);
        if (transB) {
            #pragma unroll
            for (int r0 = 0; r0 < 4; ++r0)
                rb[r0] = *(const float4*)(B + (long long)(bn + r0 * 32 + tr) * ldb + kb + tc);
        } else {
            #pragma unroll
            for (int k0 = 0; k0 < 4; ++k0)
                rb[k0] = *(const float4*)(B + (long long)(kb + k0 * 8 + tr2) * ldb + bn + tc2);
        }
    };

    // regs -> LDS (bf16 convert, packed dword stores); loadcnt wait lands here
    auto store_tiles = [&](int buf) {
        #pragma unroll
        for (int r0 = 0; r0 < 4; ++r0) {
            unsigned int* dst = (unsigned int*)&lds_a[buf][(r0 * 32 + tr) * LDSS + tc];
            dst[0] = f2bf2(ra[r0].x, ra[r0].y);
            dst[1] = f2bf2(ra[r0].z, ra[r0].w);
        }
        if (transB) {
            #pragma unroll
            for (int r0 = 0; r0 < 4; ++r0) {
                unsigned int* dst = (unsigned int*)&lds_b[buf][(r0 * 32 + tr) * LDSS + tc];
                dst[0] = f2bf2(rb[r0].x, rb[r0].y);
                dst[1] = f2bf2(rb[r0].z, rb[r0].w);
            }
        } else {
            #pragma unroll
            for (int k0 = 0; k0 < 4; ++k0) {
                const int kk = k0 * 8 + tr2;
                lds_b[buf][(tc2 + 0) * LDSS + kk] = f2bf(rb[k0].x);
                lds_b[buf][(tc2 + 1) * LDSS + kk] = f2bf(rb[k0].y);
                lds_b[buf][(tc2 + 2) * LDSS + kk] = f2bf(rb[k0].z);
                lds_b[buf][(tc2 + 3) * LDSS + kk] = f2bf(rb[k0].w);
            }
        }
    };

    floatx8 acc[4][2];
    for (int i = 0; i < 4; ++i)
        for (int j = 0; j < 2; ++j)
            for (int e = 0; e < 8; ++e) acc[i][j][e] = 0.0f;

    // ---- pipeline prologue ----
    load_tiles(0);
    store_tiles(0);
    if (32 < K) load_tiles(32);        // prefetch K-step 1 (no wait yet)
    __syncthreads();

    int buf = 0;
    for (int kb = 0; kb < K; kb += 32) {
        // ---- compute on lds[buf] ----
        bf16x16 afrag[4], bfrag[2];
        #pragma unroll
        for (int i = 0; i < 4; ++i) {
            const int m = wm + i * 16 + l16;
            const bf16x8 lo = *(const bf16x8*)&lds_a[buf][m * LDSS + g * 8];      // K = 8g..8g+7
            const bf16x8 hi = *(const bf16x8*)&lds_a[buf][m * LDSS + 16 + g * 8]; // K = 16+8g..
            afrag[i] = __builtin_shufflevector(lo, hi, 0,1,2,3,4,5,6,7,8,9,10,11,12,13,14,15);
        }
        #pragma unroll
        for (int j = 0; j < 2; ++j) {
            const int n = wn + j * 16 + l16;
            const bf16x8 lo = *(const bf16x8*)&lds_b[buf][n * LDSS + g * 8];      // K = 8g..8g+7
            const bf16x8 hi = *(const bf16x8*)&lds_b[buf][n * LDSS + 16 + g * 8]; // K = 16+8g..
            bfrag[j] = __builtin_shufflevector(lo, hi, 0,1,2,3,4,5,6,7,8,9,10,11,12,13,14,15);
        }
        #pragma unroll
        for (int i = 0; i < 4; ++i)
            #pragma unroll
            for (int j = 0; j < 2; ++j)
                acc[i][j] = __builtin_amdgcn_wmma_f32_16x16x32_bf16(
                    false, afrag[i], false, bfrag[j], (short)0, acc[i][j], false, false);

        // ---- stage next tile into the other LDS buffer; prefetch tile after ----
        if (kb + 32 < K) {
            store_tiles(buf ^ 1);              // waits on loads issued last iteration
            if (kb + 64 < K) load_tiles(kb + 64);
        }
        __syncthreads();
        buf ^= 1;
    }

    // ---- epilogue: bias (+relu), fp32 store. C elem e -> row 8g+e, col l16
    #pragma unroll
    for (int i = 0; i < 4; ++i) {
        #pragma unroll
        for (int j = 0; j < 2; ++j) {
            const int ncol = bn + wn + j * 16 + l16;
            const float bv = bias ? bias[ncol] : 0.0f;
            #pragma unroll
            for (int e = 0; e < 8; ++e) {
                const int mrow = bm + wm + i * 16 + g * 8 + e;
                float val = acc[i][j][e] + bv;
                if (relu) val = fmaxf(val, 0.0f);
                C[(long long)mrow * ldc + ncol] = val;
            }
        }
    }
}

// ---------------------------------------------------------------------------
// Softmax over rows of length 512 (in-place), pre-scaled by `scale`.
// ---------------------------------------------------------------------------
__global__ __launch_bounds__(256) void softmax512(float* __restrict__ S, float scale)
{
    __shared__ float red[256];
    float* p = S + (long long)blockIdx.x * 512;
    const int tid = threadIdx.x;
    float x0 = p[tid] * scale, x1 = p[tid + 256] * scale;
    red[tid] = fmaxf(x0, x1);
    __syncthreads();
    for (int s = 128; s > 0; s >>= 1) { if (tid < s) red[tid] = fmaxf(red[tid], red[tid + s]); __syncthreads(); }
    const float m = red[0];
    __syncthreads();
    const float e0 = expf(x0 - m), e1 = expf(x1 - m);
    red[tid] = e0 + e1;
    __syncthreads();
    for (int s = 128; s > 0; s >>= 1) { if (tid < s) red[tid] += red[tid + s]; __syncthreads(); }
    const float inv = 1.0f / red[0];
    p[tid] = e0 * inv;
    p[tid + 256] = e1 * inv;
}

// ---------------------------------------------------------------------------
// H[row] = LayerNorm(X[row] + H[row]) * g + b, D = 512 (in-place on H)
// ---------------------------------------------------------------------------
__global__ __launch_bounds__(256) void add_layernorm512(
    const float* __restrict__ X, float* __restrict__ H,
    const float* __restrict__ gam, const float* __restrict__ bet)
{
    __shared__ float red[256];
    const long long row = blockIdx.x;
    const float* xp = X + row * 512;
    float* hp = H + row * 512;
    const int tid = threadIdx.x;
    const float v0 = xp[tid] + hp[tid];
    const float v1 = xp[tid + 256] + hp[tid + 256];
    red[tid] = v0 + v1;
    __syncthreads();
    for (int s = 128; s > 0; s >>= 1) { if (tid < s) red[tid] += red[tid + s]; __syncthreads(); }
    const float mu = red[0] * (1.0f / 512.0f);
    __syncthreads();
    const float d0 = v0 - mu, d1 = v1 - mu;
    red[tid] = d0 * d0 + d1 * d1;
    __syncthreads();
    for (int s = 128; s > 0; s >>= 1) { if (tid < s) red[tid] += red[tid + s]; __syncthreads(); }
    const float rstd = rsqrtf(red[0] * (1.0f / 512.0f) + 1e-5f);
    hp[tid]       = d0 * rstd * gam[tid]       + bet[tid];
    hp[tid + 256] = d1 * rstd * gam[tid + 256] + bet[tid + 256];
}

// ---------------------------------------------------------------------------
// h[t,b,:448] = x[t,b,:] @ W_in + b_in ; h[t,b,448:] = sin/cos pos-emb(t)
// One block per row (t*B+b), K=128, N=512.
// ---------------------------------------------------------------------------
__global__ __launch_bounds__(256) void input_embed(
    const float* __restrict__ x, const float* __restrict__ Win,
    const float* __restrict__ bin, float* __restrict__ H)
{
    __shared__ float xrow[128];
    const int row = blockIdx.x;
    const int t = row >> 3;   // B = 8
    const int tid = threadIdx.x;
    if (tid < 128) xrow[tid] = x[(long long)row * 128 + tid];
    __syncthreads();
    float* hp = H + (long long)row * 512;
    for (int j = tid; j < 512; j += 256) {
        float val;
        if (j < 448) {
            float s = bin[j];
            for (int f = 0; f < 128; ++f) s += xrow[f] * Win[f * 448 + j];
            val = s;
        } else {
            const int p = j - 448;
            const int i = (p < 32) ? p : (p - 32);
            const float inv = expf(-(logf(100000.0f) / 31.0f) * (float)i);
            const float arg = (float)t * inv;
            val = (p < 32) ? sinf(arg) : cosf(arg);
        }
        hp[j] = val;
    }
}

// ---------------------------------------------------------------------------
extern "C" void kernel_launch(void* const* d_in, const int* in_sizes, int n_in,
                              void* d_out, int out_size, void* d_ws, size_t ws_size,
                              hipStream_t stream)
{
    (void)in_sizes; (void)n_in; (void)out_size; (void)ws_size;
    const float* x    = (const float*)d_in[0];
    const float* W_in = (const float*)d_in[1];
    const float* b_in = (const float*)d_in[2];
    const float* Wq   = (const float*)d_in[3];
    const float* bq   = (const float*)d_in[4];
    const float* Wk   = (const float*)d_in[5];
    const float* bk   = (const float*)d_in[6];
    const float* Wv   = (const float*)d_in[7];
    const float* bv   = (const float*)d_in[8];
    const float* Wo   = (const float*)d_in[9];
    const float* bo   = (const float*)d_in[10];
    const float* ln1g = (const float*)d_in[11];
    const float* ln1b = (const float*)d_in[12];
    const float* W1   = (const float*)d_in[13];
    const float* b1   = (const float*)d_in[14];
    const float* W2   = (const float*)d_in[15];
    const float* b2   = (const float*)d_in[16];
    const float* ln2g = (const float*)d_in[17];
    const float* ln2b = (const float*)d_in[18];

    // T=512 B=8 D=512 H=8 -> rows M=4096, HD=4096, BH=64, DFF=2048
    const long long MB = 1024LL * 1024LL;
    float* ws  = (float*)d_ws;
    float* h   = ws;               //  2M floats (4096 x 512)
    float* q   = ws + 2 * MB;      // 16M floats (512 x 64*512)
    float* kb_ = ws + 18 * MB;     // 16M floats
    float* v   = ws + 34 * MB;     // 16M floats
    float* sim = ws + 50 * MB;     // 16M floats (64 x 512 x 512)
    float* c   = q;                // alias: q dead after scores
    float* tmp = kb_;              // alias: k dead after scores (2M floats)
    float* ff1 = kb_ + 2 * MB;     // alias: 8M floats

    const dim3 blk(256);
    const float scale = 0.044194173824159216f; // 1/sqrt(512)

    input_embed<<<4096, blk, 0, stream>>>(x, W_in, b_in, h);

    for (int L = 0; L < 6; ++L) {
        const float* wq = Wq + (long long)L * 512 * 4096;
        const float* wk = Wk + (long long)L * 512 * 4096;
        const float* wv = Wv + (long long)L * 512 * 4096;
        const float* wo = Wo + (long long)L * 4096 * 512;
        const float* w1 = W1 + (long long)L * 512 * 2048;
        const float* w2 = W2 + (long long)L * 2048 * 512;

        // Q/K/V = h @ W + b : (4096 x 512) x (512 x 4096)
        gemm_wmma_bf16<<<dim3(32, 32, 1), blk, 0, stream>>>(
            h, wq, bq + (long long)L * 4096, q, 4096, 4096, 512, 512, 4096, 4096, 0, 0, 0, 0, 0);
        gemm_wmma_bf16<<<dim3(32, 32, 1), blk, 0, stream>>>(
            h, wk, bk + (long long)L * 4096, kb_, 4096, 4096, 512, 512, 4096, 4096, 0, 0, 0, 0, 0);
        gemm_wmma_bf16<<<dim3(32, 32, 1), blk, 0, stream>>>(
            h, wv, bv + (long long)L * 4096, v, 4096, 4096, 512, 512, 4096, 4096, 0, 0, 0, 0, 0);

        // sim[bh] = Q_bh (512x512, row-stride 32768) @ K_bh^T  (NT, batched over 64)
        gemm_wmma_bf16<<<dim3(4, 4, 64), blk, 0, stream>>>(
            q, kb_, nullptr, sim, 512, 512, 512, 32768, 32768, 512,
            512, 512, 512LL * 512LL, 1, 0);

        softmax512<<<64 * 512, blk, 0, stream>>>(sim, scale);

        // c[bh] = P_bh (512x512) @ V_bh (512x512, row-stride 32768)  (NN, batched)
        gemm_wmma_bf16<<<dim3(4, 4, 64), blk, 0, stream>>>(
            sim, v, nullptr, c, 512, 512, 512, 512, 32768, 32768,
            512LL * 512LL, 512, 512, 0, 0);

        // tmp = c @ Wo + bo : (4096 x 4096) x (4096 x 512)
        gemm_wmma_bf16<<<dim3(4, 32, 1), blk, 0, stream>>>(
            c, wo, bo + (long long)L * 512, tmp, 4096, 512, 4096, 4096, 512, 512, 0, 0, 0, 0, 0);

        add_layernorm512<<<4096, blk, 0, stream>>>(tmp, h, ln1g + (long long)L * 512, ln1b + (long long)L * 512);

        // ff1 = relu(h @ W1 + b1) : (4096 x 512) x (512 x 2048)
        gemm_wmma_bf16<<<dim3(16, 32, 1), blk, 0, stream>>>(
            h, w1, b1 + (long long)L * 2048, ff1, 4096, 2048, 512, 512, 2048, 2048, 0, 0, 0, 0, 1);

        // tmp = ff1 @ W2 + b2 : (4096 x 2048) x (2048 x 512)
        gemm_wmma_bf16<<<dim3(4, 32, 1), blk, 0, stream>>>(
            ff1, w2, b2 + (long long)L * 512, tmp, 4096, 512, 2048, 2048, 512, 512, 0, 0, 0, 0, 0);

        add_layernorm512<<<4096, blk, 0, stream>>>(tmp, h, ln2g + (long long)L * 512, ln2b + (long long)L * 512);
    }

    hipMemcpyAsync(d_out, h, (size_t)4096 * 512 * sizeof(float),
                   hipMemcpyDeviceToDevice, stream);
}
